// RepSSA_3985729650872
// MI455X (gfx1250) — matrix operand order repaired and checked by reading, not compiled, not run on hardware
//
#include <hip/hip_runtime.h>
#include <stdint.h>

// ---------------------------------------------------------------------------
// RepSSA (spiking linear attention) for MI455X / gfx1250, wave32 + WMMA.
// Shapes fixed by the reference: T=4, B=16, C=512, N=H*W=1024, h=8, d=64.
//
// Pipeline (all kernels on `stream`, serialized by stream order):
//   convQ -> lifQ -> convK -> lifK -> convV -> lifV      (bf16 WMMA)
//   kv = k^T v            (f16 WMMA, exact: spikes are {0,1})
//   out = q @ kv * 0.125  (f16 WMMA, exact; stored [t][b][n][c], coalesced)
//   lif(out) -> proj conv (f16 WMMA, B tile async-copied to LDS) ->
//   conn_lif(proj, x) -> d_out
//
// Workspace layout (bytes), total ~390 MiB:
//   S0  f32 scratch                : 134217728   conv/proj: [t][b][c][n]; attn: [t][b][n][c]
//   Sq  q spikes f16               :  67108864   [t][b][h][n][d]   (k-contig A operand)
//   Sk  k spikes f16               :  67108864   [t][b][c][n]
//   Sv  v spikes f16               :  67108864   [t][b][c][n]
//   Sa  attn spikes f16            :  67108864   [t][b][n][c]      (k-contig B operand)
//   Skv kv^T f16                   :   4194304   [t][b][h][e][d]   (k-contig B operand)
// ---------------------------------------------------------------------------

typedef __attribute__((ext_vector_type(16))) _Float16 v16h;
typedef __attribute__((ext_vector_type(8)))  _Float16 v8h;
typedef __attribute__((ext_vector_type(16))) __bf16   v16bf;
typedef __attribute__((ext_vector_type(8)))  __bf16   v8bf;
typedef __attribute__((ext_vector_type(8)))  float    v8f;
typedef __attribute__((ext_vector_type(4)))  int      v4i;

template <typename V8>
__device__ __forceinline__ auto combine16(V8 lo, V8 hi) {
  return __builtin_shufflevector(lo, hi, 0,1,2,3,4,5,6,7,8,9,10,11,12,13,14,15);
}

__device__ __forceinline__ v8f wmma_mma(v16bf a, v16bf b, v8f c) {
  return __builtin_amdgcn_wmma_f32_16x16x32_bf16(false, a, false, b, (short)0, c, false, false);
}
__device__ __forceinline__ v8f wmma_mma(v16h a, v16h b, v8f c) {
  return __builtin_amdgcn_wmma_f32_16x16x32_f16(false, a, false, b, (short)0, c, false, false);
}

template <typename T> struct vec16;
template <> struct vec16<__bf16>   { using t = v16bf; using h = v8bf; };
template <> struct vec16<_Float16> { using t = v16h;  using h = v8h;  };

#define C_DIM 512
#define N_DIM 1024
#define TOTAL (16 * C_DIM * N_DIM)   // per-t element count B*C*N = 8388608

// --- CDNA5 async global->LDS (raw byte copy), guarded so compile can't break.
#if defined(__HIP_DEVICE_COMPILE__) && __has_builtin(__builtin_amdgcn_global_load_async_to_lds_b128)
#define HAVE_ASYNC_LDS 1
#else
#define HAVE_ASYNC_LDS 0
#endif

__device__ __forceinline__ void copy_b128_to_lds(const void* g, void* l) {
#if HAVE_ASYNC_LDS
  // prototype: (v4i addrspace(1)* src, v4i addrspace(3)* dst, imm offset, imm cpol)
  __builtin_amdgcn_global_load_async_to_lds_b128(
      (__attribute__((address_space(1))) v4i*)(uintptr_t)g,
      (__attribute__((address_space(3))) v4i*)(uintptr_t)l, 0, 0);
#else
  *(int4*)l = *(const int4*)g;
#endif
}
__device__ __forceinline__ void async_fence() {
#if HAVE_ASYNC_LDS
  asm volatile("s_wait_asynccnt 0" ::: "memory");
#endif
}

// ---------------------------------------------------------------------------
// conv1x1 as GEMM:  Y[o][n] = sum_c W[o][c] * X[..] + bias[o], per image tb.
// Workgroup tile 128(o) x 128(n), 8 waves, each wave 32x64 = 2x4 WMMA tiles.
// Double-buffered LDS: stage K-tile ks+1 while WMMA-ing tile ks.
//   BNMAJOR=false: X is [c][n] (f32/f16), transposed+converted into Bs[n][k].
//   BNMAJOR=true : X is [n][c] (f16), raw row copy (async global->LDS path).
// ---------------------------------------------------------------------------
template <typename TIN, typename T16, bool BNMAJOR>
__global__ __launch_bounds__(256) void conv_gemm(
    const float* __restrict__ W, const TIN* __restrict__ X,
    const float* __restrict__ bias, float* __restrict__ Y) {
  using V16 = typename vec16<T16>::t;
  using V8  = typename vec16<T16>::h;

  __shared__ T16 As[2][128][32];   // As[buf][o][k]
  __shared__ T16 Bs[2][128][32];   // Bs[buf][n][k]  (k contig per row -> 32B B frag)

  const int tid = threadIdx.x;
  const int tb  = blockIdx.z;                 // t*B + b
  const int c0  = blockIdx.y * 128;           // output-channel tile
  const int n0  = blockIdx.x * 128;           // spatial tile
  const TIN*  Xtb = X + (size_t)tb * C_DIM * N_DIM;
  float*      Ytb = Y + (size_t)tb * C_DIM * N_DIM;

  const int wave = tid >> 5;
  const int lane = tid & 31;
  const int wm = wave & 3;                    // 4 M-chunks of 32 rows
  const int wn = wave >> 2;                   // 2 N-chunks of 64 cols

  v8f acc[2][4] = {};

  const int fm  = lane & 15;                  // M (A) / N (B) index inside tile
  const int kb  = (lane < 16) ? 0 : 8;        // unused: A frag also reads K half {kh..kh+15}
  const int kh  = (lane < 16) ? 0 : 16;       // B frag: K half {kh..kh+15}

  auto stage = [&](int buf, int k0) {
    // A tile: W[c0+row][k0+col] -> T16
#pragma unroll
    for (int i = 0; i < 16; ++i) {
      int idx = tid + i * 256;
      int row = idx >> 5, col = idx & 31;
      As[buf][row][col] = (T16)W[(size_t)(c0 + row) * C_DIM + (k0 + col)];
    }
    if constexpr (BNMAJOR) {
      // raw 16B row-chunk copies, 2 per thread (128 rows x 64B)
#pragma unroll
      for (int i = 0; i < 2; ++i) {
        int idx = tid + i * 256;                 // 0..511
        int n = idx >> 2, ch = idx & 3;
        copy_b128_to_lds(&Xtb[(size_t)(n0 + n) * C_DIM + k0 + ch * 8],
                         &Bs[buf][n][ch * 8]);
      }
    } else {
      // transpose + convert: X[k0+kk][n0+n] -> Bs[n][kk]
#pragma unroll
      for (int i = 0; i < 16; ++i) {
        int idx = tid + i * 256;
        int n = idx & 127, kk = idx >> 7;
        Bs[buf][n][kk] = (T16)(float)Xtb[(size_t)(k0 + kk) * N_DIM + (n0 + n)];
      }
      if (k0 + 32 < C_DIM)   // prefetch next K-tile (global_prefetch_b8)
        __builtin_prefetch(&Xtb[(size_t)(k0 + 32) * N_DIM + n0 + (tid & 127)], 0, 1);
    }
  };

  stage(0, 0);
  async_fence();
  __syncthreads();

  const int NK = C_DIM / 32;
  for (int ks = 0; ks < NK; ++ks) {
    const int cur = ks & 1;
    if (ks + 1 < NK) stage(cur ^ 1, (ks + 1) * 32);

    V16 a[2], b[4];
#pragma unroll
    for (int mt = 0; mt < 2; ++mt) {
      const T16* p = &As[cur][wm * 32 + mt * 16 + fm][0];
      a[mt] = combine16(*(const V8*)(p + kh), *(const V8*)(p + kh + 8));
    }
#pragma unroll
    for (int nt = 0; nt < 4; ++nt)
      b[nt] = *(const V16*)&Bs[cur][wn * 64 + nt * 16 + fm][kh];
#pragma unroll
    for (int mt = 0; mt < 2; ++mt)
#pragma unroll
      for (int nt = 0; nt < 4; ++nt)
        acc[mt][nt] = wmma_mma(a[mt], b[nt], acc[mt][nt]);

    async_fence();
    __syncthreads();
  }

  // store: f32 C/D layout: VGPR r, lanes 0-15 -> M=r, lanes 16-31 -> M=r+8
  const int nl = lane & 15;
  const int mo = (lane < 16) ? 0 : 8;
#pragma unroll
  for (int mt = 0; mt < 2; ++mt)
#pragma unroll
    for (int nt = 0; nt < 4; ++nt)
#pragma unroll
      for (int r = 0; r < 8; ++r) {
        int o = c0 + wm * 32 + mt * 16 + mo + r;
        int n = n0 + wn * 64 + nt * 16 + nl;
        Ytb[(size_t)o * N_DIM + n] = acc[mt][nt][r] + bias[o];
      }
}

// ---------------------------------------------------------------------------
// LIF over T=4 (v = 0.5 v + 0.5 x; spike = v>=1; hard reset), writes f16 spikes.
// layoutQ=1: input [t][b][c][n], output [t][b][h][n][d] (q A-operand layout).
// layoutQ=0: pure elementwise pass-through of the flat index per t.
// ---------------------------------------------------------------------------
__global__ __launch_bounds__(256) void lif_spikes(
    const float* __restrict__ Cv, _Float16* __restrict__ Sp, int layoutQ) {
  int idx = blockIdx.x * blockDim.x + threadIdx.x;
  if (idx >= TOTAL) return;
  const int n = idx & (N_DIM - 1);
  const int c = (idx >> 10) & (C_DIM - 1);
  const int b = idx >> 19;
  float v = 0.f;
#pragma unroll
  for (int t = 0; t < 4; ++t) {
    float x = Cv[(size_t)t * TOTAL + idx];
    v = 0.5f * (v + x);
    float s = (v >= 1.0f) ? 1.f : 0.f;
    if (layoutQ) {
      int hh = c >> 6, dd = c & 63;
      Sp[((((size_t)t * 16 + b) * 8 + hh) * N_DIM + n) * 64 + dd] = (_Float16)s;
    } else {
      Sp[(size_t)t * TOTAL + idx] = (_Float16)s;
    }
    v *= (1.f - s);
  }
}

// ---------------------------------------------------------------------------
// kv = k^T v per (t,b,h): 64x64 output, K = N = 1024. One workgroup per tbh.
// 8 waves x 2 WMMA tiles (sharing the A fragment). Stores kv^T as f16
// (values are integers <= 1024, exact in f16) at [e][d] for the next GEMM.
// ---------------------------------------------------------------------------
__global__ __launch_bounds__(256) void kv_gemm(
    const _Float16* __restrict__ Ksp, const _Float16* __restrict__ Vsp,
    _Float16* __restrict__ KvT) {
  const int tbh = blockIdx.x;            // (t*B+b)*8 + h
  const int tb = tbh >> 3, hh = tbh & 7;
  const _Float16* kb_ = Ksp + ((size_t)tb * C_DIM + hh * 64) * N_DIM;
  const _Float16* vb_ = Vsp + ((size_t)tb * C_DIM + hh * 64) * N_DIM;

  const int wave = threadIdx.x >> 5, lane = threadIdx.x & 31;
  const int mt = wave >> 1;              // 4 M-tiles of kv
  const int nt0 = (wave & 1) * 2;        // 2 consecutive N-tiles per wave
  const int fm = lane & 15;
  const int kbo = (lane < 16) ? 0 : 8;
  const int kh  = (lane < 16) ? 0 : 16;

  v8f acc0 = {}, acc1 = {};
  for (int ks = 0; ks < N_DIM; ks += 32) {
    const _Float16* pa = kb_ + (size_t)(mt * 16 + fm) * N_DIM + ks;
    v16h a = combine16(*(const v8h*)(pa + kh), *(const v8h*)(pa + kh + 8));
    v16h b0 = *(const v16h*)(vb_ + (size_t)((nt0 + 0) * 16 + fm) * N_DIM + ks + kh);
    v16h b1 = *(const v16h*)(vb_ + (size_t)((nt0 + 1) * 16 + fm) * N_DIM + ks + kh);
    acc0 = wmma_mma(a, b0, acc0);
    acc1 = wmma_mma(a, b1, acc1);
  }

  _Float16* kvp = KvT + (size_t)tbh * 64 * 64;
  const int mo = (lane < 16) ? 0 : 8;
  const int e0 = nt0 * 16 + (lane & 15);
#pragma unroll
  for (int r = 0; r < 8; ++r) {
    int dd = mt * 16 + mo + r;
    kvp[(size_t)(e0)*64 + dd]        = (_Float16)acc0[r];
    kvp[(size_t)(e0 + 16) * 64 + dd] = (_Float16)acc1[r];
  }
}

// ---------------------------------------------------------------------------
// out = q @ kv * SCALE per (t,b,h): M = 1024 (spatial), N = 64, K = 64.
// grid = (tbh, m-chunk of 128 rows); 8 waves x 16 rows x 4 N-tiles each.
// Writes f32 into S0 in [t][b][n][c] layout -> contiguous 64B stores per row.
// ---------------------------------------------------------------------------
__global__ __launch_bounds__(256) void attn_out(
    const _Float16* __restrict__ Qsp, const _Float16* __restrict__ KvT,
    float* __restrict__ Y) {
  const int tbh = blockIdx.x;
  const int tb = tbh >> 3, hh = tbh & 7;
  const _Float16* qb  = Qsp + (size_t)tbh * N_DIM * 64;
  const _Float16* kvp = KvT + (size_t)tbh * 64 * 64;
  float* yb = Y + (size_t)tb * C_DIM * N_DIM;   // [n][c] layout per tb

  const int wave = threadIdx.x >> 5, lane = threadIdx.x & 31;
  const int m0 = blockIdx.y * 128 + wave * 16;
  const int fm = lane & 15;
  const int kbo = (lane < 16) ? 0 : 8;
  const int kh  = (lane < 16) ? 0 : 16;

  v8f acc[4] = {};
#pragma unroll
  for (int ks = 0; ks < 64; ks += 32) {
    const _Float16* pa = qb + (size_t)(m0 + fm) * 64 + ks;
    v16h a = combine16(*(const v8h*)(pa + kh), *(const v8h*)(pa + kh + 8));
#pragma unroll
    for (int nt = 0; nt < 4; ++nt) {
      v16h b = *(const v16h*)(kvp + (size_t)(nt * 16 + fm) * 64 + ks + kh);
      acc[nt] = wmma_mma(a, b, acc[nt]);
    }
  }

  const int mo = (lane < 16) ? 0 : 8;
#pragma unroll
  for (int nt = 0; nt < 4; ++nt)
#pragma unroll
    for (int r = 0; r < 8; ++r) {
      int n = m0 + mo + r;
      int e = nt * 16 + (lane & 15);
      yb[(size_t)n * C_DIM + hh * 64 + e] = acc[nt][r] * 0.125f;
    }
}

// ---------------------------------------------------------------------------
// conn_lif: v = 0.5 v + 0.5 (proj + x); spike; hard reset. Writes f32 d_out.
// ---------------------------------------------------------------------------
__global__ __launch_bounds__(256) void conn_lif(
    const float* __restrict__ P, const float* __restrict__ X,
    float* __restrict__ Out) {
  int idx = blockIdx.x * blockDim.x + threadIdx.x;
  if (idx >= TOTAL) return;
  float v = 0.f;
#pragma unroll
  for (int t = 0; t < 4; ++t) {
    float u = P[(size_t)t * TOTAL + idx] + X[(size_t)t * TOTAL + idx];
    v = 0.5f * (v + u);
    float s = (v >= 1.0f) ? 1.f : 0.f;
    Out[(size_t)t * TOTAL + idx] = s;
    v *= (1.f - s);
  }
}

// ---------------------------------------------------------------------------
extern "C" void kernel_launch(void* const* d_in, const int* in_sizes, int n_in,
                              void* d_out, int out_size, void* d_ws, size_t ws_size,
                              hipStream_t stream) {
  const float* x  = (const float*)d_in[0];
  const float* Wq = (const float*)d_in[1];
  const float* bq = (const float*)d_in[2];
  const float* Wk = (const float*)d_in[3];
  const float* bk = (const float*)d_in[4];
  const float* Wv = (const float*)d_in[5];
  const float* bv = (const float*)d_in[6];
  const float* Wp = (const float*)d_in[7];
  const float* bp = (const float*)d_in[8];
  // d_in[9] = num_heads (fixed at 8 here)

  char* ws = (char*)d_ws;
  float*     S0  = (float*)ws;                            // 134217728 B
  _Float16*  Sq  = (_Float16*)(ws + 134217728ull);        //  67108864 B
  _Float16*  Sk  = (_Float16*)(ws + 201326592ull);
  _Float16*  Sv  = (_Float16*)(ws + 268435456ull);
  _Float16*  Sa  = (_Float16*)(ws + 335544320ull);
  _Float16*  Skv = (_Float16*)(ws + 402653184ull);        //   4194304 B

  dim3 b256(256);
  dim3 gconv(N_DIM / 128, C_DIM / 128, 64);               // (8, 4, T*B)
  dim3 gelem((TOTAL + 255) / 256);

  // Q/K/V branches: conv (bf16 WMMA) then LIF -> f16 spikes
  conv_gemm<float, __bf16, false><<<gconv, b256, 0, stream>>>(Wq, x, bq, S0);
  lif_spikes<<<gelem, b256, 0, stream>>>(S0, Sq, 1);
  conv_gemm<float, __bf16, false><<<gconv, b256, 0, stream>>>(Wk, x, bk, S0);
  lif_spikes<<<gelem, b256, 0, stream>>>(S0, Sk, 0);
  conv_gemm<float, __bf16, false><<<gconv, b256, 0, stream>>>(Wv, x, bv, S0);
  lif_spikes<<<gelem, b256, 0, stream>>>(S0, Sv, 0);

  // linear attention (exact in f16: spike operands are {0,1})
  kv_gemm<<<dim3(512), b256, 0, stream>>>(Sk, Sv, Skv);
  attn_out<<<dim3(512, 8), b256, 0, stream>>>(Sq, Skv, S0);   // S0 now [t][b][n][c]

  // attention LIF (elementwise, keeps [n][c] layout), projection conv with
  // n-major f16 B operand (raw async LDS copies), connecting LIF
  lif_spikes<<<gelem, b256, 0, stream>>>(S0, Sa, 0);
  conv_gemm<_Float16, _Float16, true><<<gconv, b256, 0, stream>>>(Wp, Sa, bp, S0);
  conn_lif<<<gelem, b256, 0, stream>>>(S0, x, (float*)d_out);
}
